// SparseGCNConv_89507118448761
// MI455X (gfx1250) — matrix operand order repaired and checked by reading, not compiled, not run on hardware
//
#include <hip/hip_runtime.h>

typedef __attribute__((ext_vector_type(16))) _Float16     v16h;
typedef __attribute__((ext_vector_type(8)))  float        v8f;
typedef __attribute__((ext_vector_type(8)))  unsigned int v8u;

#define FIN 512
#define FOUT 128
#define AT_STRIDE 516   // 512 + 4 pad -> 16 row-lanes hit 16 distinct LDS banks

// ---------------- utility kernels ----------------
__global__ void zero_i32_k(int* __restrict__ p, int n) {
  int i = blockIdx.x * 256 + threadIdx.x;
  if (i < n) p[i] = 0;
}
__global__ void zero_f32_k(float* __restrict__ p, int n) {
  int i = blockIdx.x * 256 + threadIdx.x;
  if (i < n) p[i] = 0.0f;
}
__global__ void count_k(const int* __restrict__ rows, int* __restrict__ counts, int nnz) {
  int i = blockIdx.x * 256 + threadIdx.x;
  if (i < nnz) atomicAdd(&counts[rows[i]], 1);
}

// single-workgroup chunked exclusive scan (100k elements -> trivial work)
__global__ __launch_bounds__(1024) void scan_k(const int* __restrict__ counts,
                                               int* __restrict__ ofs,
                                               int* __restrict__ cursor, int n) {
  __shared__ int sh[1024];
  __shared__ int carry_s;
  if (threadIdx.x == 0) carry_s = 0;
  __syncthreads();
  for (int base = 0; base < n; base += 1024) {
    int i = base + (int)threadIdx.x;
    int v = (i < n) ? counts[i] : 0;
    sh[threadIdx.x] = v;
    __syncthreads();
    for (int off = 1; off < 1024; off <<= 1) {
      int t = (threadIdx.x >= (unsigned)off) ? sh[threadIdx.x - off] : 0;
      __syncthreads();
      sh[threadIdx.x] += t;
      __syncthreads();
    }
    int excl = sh[threadIdx.x] - v + carry_s;
    if (i < n) { ofs[i] = excl; cursor[i] = excl; }
    __syncthreads();
    if (threadIdx.x == 1023) carry_s += sh[1023];
    __syncthreads();
  }
  if (threadIdx.x == 0) ofs[n] = carry_s;
}

__global__ void fill_k(const int* __restrict__ rows, const int* __restrict__ cols,
                       const float* __restrict__ vals, int* __restrict__ cursor,
                       int* __restrict__ csr_col, float* __restrict__ csr_val, int nnz) {
  int i = blockIdx.x * 256 + threadIdx.x;
  if (i >= nnz) return;
  int pos = atomicAdd(&cursor[rows[i]], 1);
  csr_col[pos] = cols[i];
  csr_val[pos] = vals[i];
}

// pack f32 weight into B-fragment-native layout:
// Wpk[k/2][col] = { f16(W[k][col]), f16(W[k+1][col]) }  (one dword = one B VGPR)
__global__ void wpack_k(const float* __restrict__ w, unsigned int* __restrict__ wpk, int npk) {
  int i = blockIdx.x * 256 + threadIdx.x;
  if (i >= npk) return;
  int pk = i / FOUT, col = i - pk * FOUT;
  _Float16 lo = (_Float16)w[(2 * pk) * FOUT + col];
  _Float16 hi = (_Float16)w[(2 * pk + 1) * FOUT + col];
  unsigned int ulo = (unsigned int)__builtin_bit_cast(unsigned short, lo);
  unsigned int uhi = (unsigned int)__builtin_bit_cast(unsigned short, hi);
  wpk[pk * FOUT + col] = ulo | (uhi << 16);
}

// ---------------- phase 1: densify row-tile in LDS + WMMA GEMM ----------------
// one block = 16 node rows; 8 waves each own a 16-wide slice of the 128 outputs
__global__ __launch_bounds__(256) void gemm_wmma_k(const int* __restrict__ row_ofs,
                                                   const int* __restrict__ csr_col,
                                                   const float* __restrict__ csr_val,
                                                   const unsigned int* __restrict__ Wpk,
                                                   float* __restrict__ base_out,
                                                   int n_nodes) {
  __shared__ float At[16 * AT_STRIDE];  // f32 accumulation tile (handles duplicate nnz)
  __shared__ int ro[17];
  const int tid = threadIdx.x;
  const int r0 = blockIdx.x * 16;

  for (int i = tid; i < 16 * AT_STRIDE; i += 256) At[i] = 0.0f;
  if (tid < 17) {
    int r = r0 + tid;
    ro[tid] = row_ofs[r > n_nodes ? n_nodes : r];
  }
  __syncthreads();

  // CSR entries for 16 consecutive rows are contiguous: scatter into LDS tile
  const int s = ro[0], e = ro[16];
  for (int idx = s + tid; idx < e; idx += 256) {
    int j = 0;                         // find local row via 4-step binary search
    if (ro[j + 8] <= idx) j += 8;
    if (ro[j + 4] <= idx) j += 4;
    if (ro[j + 2] <= idx) j += 2;
    if (ro[j + 1] <= idx) j += 1;
    atomicAdd(&At[j * AT_STRIDE + csr_col[idx]], csr_val[idx]);   // ds_add_f32
  }
  __syncthreads();

  const int wave = tid >> 5;           // N-tile index 0..7
  const int lane = tid & 31;
  const int m    = lane & 15;
  const int hi   = lane >> 4;
  const int col  = wave * 16 + m;

  v8f acc = {};
#pragma unroll 2
  for (int k0 = 0; k0 < FIN; k0 += 32) {
    v16h a;
#pragma unroll
    for (int i = 0; i < 8; ++i) {      // A 16x32 f16 layout (ISA 7.12.2)
      const int K = 2 * i + (i >= 4 ? 8 : 0) + hi * 8;
      a[2 * i]     = (_Float16)At[m * AT_STRIDE + k0 + K];
      a[2 * i + 1] = (_Float16)At[m * AT_STRIDE + k0 + K + 1];
    }
    v8u u;                             // B 32x16: VGPR i = K-pair (2i+(i>=4?8:0)+8*hi, +1)
#pragma unroll
    for (int i = 0; i < 8; ++i)
      u[i] = Wpk[((k0 >> 1) + i + (i >= 4 ? 4 : 0) + hi * 4) * FOUT + col];
    v16h b = __builtin_bit_cast(v16h, u);
    acc = __builtin_amdgcn_wmma_f32_16x16x32_f16(false, a, false, b,
                                                 (short)0, acc, false, false);
  }

  // C/D layout: VGPR v -> M = v + 8*hi, N = lane%16
  float* bp = base_out + (size_t)(r0 + hi * 8) * FOUT + col;
  if (r0 + 16 <= n_nodes) {            // uniform fast path: no per-store predication
#pragma unroll
    for (int v = 0; v < 8; ++v) bp[(size_t)v * FOUT] = acc[v];
  } else {
#pragma unroll
    for (int v = 0; v < 8; ++v)
      if (r0 + hi * 8 + v < n_nodes) bp[(size_t)v * FOUT] = acc[v];
  }
}

// ---------------- phase 2: edge scatter (L2-resident gather + f32 atomics) ----
__global__ void scatter_k(const int* __restrict__ adj_idx, const float* __restrict__ adj_val,
                          const float* __restrict__ base, float* __restrict__ out,
                          int n_edges) {
  unsigned gid = blockIdx.x * 256u + threadIdx.x;
  int e = (int)(gid >> 5);
  int g = (int)(gid & 31u);
  if (e >= n_edges) return;
  int dst  = adj_idx[e];               // adj_indices[0] = segment (output) rows
  int src  = adj_idx[n_edges + e];     // adj_indices[1] = gathered rows
  float v  = adj_val[e];
  const float4 x = ((const float4*)(base + (size_t)src * FOUT))[g];
  float* o = out + (size_t)dst * FOUT + g * 4;
  atomicAdd(o + 0, v * x.x);
  atomicAdd(o + 1, v * x.y);
  atomicAdd(o + 2, v * x.z);
  atomicAdd(o + 3, v * x.w);
}

extern "C" void kernel_launch(void* const* d_in, const int* in_sizes, int n_in,
                              void* d_out, int out_size, void* d_ws, size_t ws_size,
                              hipStream_t stream) {
  (void)n_in; (void)ws_size;
  const int*   adj_idx = (const int*)d_in[0];
  const float* adj_val = (const float*)d_in[1];
  const int*   f_rows  = (const int*)d_in[2];
  const int*   f_cols  = (const int*)d_in[3];
  const float* f_vals  = (const float*)d_in[4];
  const float* weight  = (const float*)d_in[5];
  const int n_edges = in_sizes[0] / 2;
  const int fnnz    = in_sizes[2];
  const int N       = out_size / FOUT;
  float* out = (float*)d_out;

  char* w = (char*)d_ws;
  size_t off = 0;
  auto carve = [&](size_t bytes) -> char* {
    char* p = w + off;
    off = (off + bytes + 255) & ~(size_t)255;
    return p;
  };
  float*        base    = (float*)        carve((size_t)N * FOUT * sizeof(float));
  unsigned int* wpk     = (unsigned int*) carve((size_t)(FIN / 2) * FOUT * sizeof(unsigned int));
  int*          counts  = (int*)          carve((size_t)N * sizeof(int));
  int*          row_ofs = (int*)          carve((size_t)(N + 1) * sizeof(int));
  int*          cursor  = (int*)          carve((size_t)N * sizeof(int));
  int*          csr_col = (int*)          carve((size_t)fnnz * sizeof(int));
  float*        csr_val = (float*)        carve((size_t)fnnz * sizeof(float));

  // CSR build for sparse features
  zero_i32_k<<<(N + 255) / 256, 256, 0, stream>>>(counts, N);
  count_k<<<(fnnz + 255) / 256, 256, 0, stream>>>(f_rows, counts, fnnz);
  scan_k<<<1, 1024, 0, stream>>>(counts, row_ofs, cursor, N);
  fill_k<<<(fnnz + 255) / 256, 256, 0, stream>>>(f_rows, f_cols, f_vals,
                                                 cursor, csr_col, csr_val, fnnz);
  // phase 1: WMMA GEMM over LDS-densified row tiles
  const int npk = (FIN / 2) * FOUT;
  wpack_k<<<(npk + 255) / 256, 256, 0, stream>>>(weight, wpk, npk);
  gemm_wmma_k<<<(N + 15) / 16, 256, 0, stream>>>(row_ofs, csr_col, csr_val, wpk, base, N);
  // phase 2: neighbor aggregation
  zero_f32_k<<<(out_size + 255) / 256, 256, 0, stream>>>(out, out_size);
  scatter_k<<<(n_edges * 32 + 255) / 256, 256, 0, stream>>>(adj_idx, adj_val, base, out, n_edges);
}
